// GCN_22849226015440
// MI455X (gfx1250) — matrix operand order repaired and checked by reading, not compiled, not run on hardware
//
#include <hip/hip_runtime.h>
#include <hip/hip_bf16.h>

#define NN 100000
#define EE 1600000
#define GG 512
#define FH 128
#define ROWS 80                 // rows per GEMM block; 100000/80 = 1250 exact

typedef __bf16 bf16_t;
typedef __attribute__((ext_vector_type(16))) __bf16 v16bf;
typedef __attribute__((ext_vector_type(8)))  float  v8f;

// ---------------------------------------------------------------- utilities
__global__ void k_fill(float* __restrict__ p, long n, float v) {
    long i = (long)blockIdx.x * blockDim.x + threadIdx.x;
    long stride = (long)gridDim.x * blockDim.x;
    for (; i < n; i += stride) p[i] = v;
}

// f32 [k][n] -> bf16 in WMMA-B fragment-major order:
//   dst = ((kt*128 + n)*2 + lanehalf)*16 + e,  k = kt*32 + lanehalf*16 + e
// so each lane's 16 B-elements are contiguous (2x ds_load_b128 in the GEMM).
__global__ void k_convert_w(const float* __restrict__ W, bf16_t* __restrict__ Wb) {
    int i = blockIdx.x * blockDim.x + threadIdx.x;
    if (i >= 128 * 128) return;
    int k  = i >> 7, n = i & 127;
    int kt = k >> 5, hh = (k >> 4) & 1, e = k & 15;
    int dst = ((kt * 128 + n) * 2 + hh) * 16 + e;
    Wb[dst] = (bf16_t)W[i];
}

__global__ void k_deg(const int* __restrict__ dst, float* __restrict__ deg) {
    int e = blockIdx.x * blockDim.x + threadIdx.x;
    if (e < EE) atomicAdd(&deg[dst[e]], 1.0f);
}

__global__ void k_dinv(const float* __restrict__ deg, float* __restrict__ dinv) {
    int i = blockIdx.x * blockDim.x + threadIdx.x;
    if (i < NN) dinv[i] = rsqrtf(deg[i]);
}

// ---------------------------------------------------------------- WMMA GEMM
// Y[N,128] = cvt_bf16(opt_BN(X[N,128])) @ W[128,128]  (f32 accumulate)
// 256 threads = 8 waves; block tile ROWSx128; wave w -> col tile w, 5 row tiles.
__global__ void __launch_bounds__(256)
k_gemm_bf16(const float* __restrict__ X, const bf16_t* __restrict__ Wb,
            float* __restrict__ Y,
            const float* __restrict__ scale, const float* __restrict__ shift)
{
    __shared__ __align__(16) bf16_t sW[128 * 128];    // fragment-major, 32 KB
    __shared__ __align__(16) bf16_t sA[ROWS * 128];   // [r][k], 20 KB

    const int t    = threadIdx.x;
    const int row0 = blockIdx.x * ROWS;

    // stage W: straight contiguous copy (already fragment-swizzled in global)
    {
        const uint4* wsrc = (const uint4*)Wb;
        uint4*       wdst = (uint4*)sW;
        #pragma unroll
        for (int i = 0; i < 8; ++i) wdst[t + i * 256] = wsrc[t + i * 256];
    }
    // stage A tile, fused BN apply, f32 -> bf16
    for (int i = t; i < ROWS * 128; i += 256) {
        int r = i >> 7, k = i & 127;
        float v = X[(size_t)(row0 + r) * FH + k];
        if (scale) v = v * scale[k] + shift[k];
        sA[i] = (bf16_t)v;
    }
    __syncthreads();

    const int lane  = t & 31;
    const int lhalf = lane >> 4;       // lane half selects K sub-range
    const int lm    = lane & 15;       // row (A) / col (B,D) within tile
    const int col0  = (t >> 5) * 16;   // wave id * 16

    v8f c[5] = {};
    #pragma unroll
    for (int kt = 0; kt < 4; ++kt) {
        const int k0 = kt * 32;
        // B fragment: 16 contiguous bf16 in fragment-major sW
        v16bf b;
        const int bbase = ((kt * 128 + col0 + lm) * 2 + lhalf) * 16;
        #pragma unroll
        for (int e = 0; e < 16; ++e) b[e] = sW[bbase + e];

        #pragma unroll
        for (int rt = 0; rt < 5; ++rt) {
            v16bf a;
            const int arow = (rt * 16 + lm) * 128;
            #pragma unroll
            for (int e = 0; e < 16; ++e) {
                // A 16x32 bf16: elem e -> K = lanehalf*16 + e
                int ka = k0 + (lhalf << 4) + e;
                a[e] = sA[arow + ka];
            }
            c[rt] = __builtin_amdgcn_wmma_f32_16x16x32_bf16(
                        false, a, false, b, (short)0, c[rt], false, false);
        }
    }
    // D 16x16 f32: lane=col, VGPR r -> row r + 8*lanehalf
    #pragma unroll
    for (int rt = 0; rt < 5; ++rt) {
        #pragma unroll
        for (int r = 0; r < 8; ++r) {
            int row = row0 + rt * 16 + r + (lhalf << 3);
            Y[(size_t)row * FH + col0 + lm] = c[rt][r];
        }
    }
}

// ------------------------------------------------------- edge scatter-gather
// one wave per edge; lane l moves features [4l, 4l+4) -> coalesced float4
__global__ void __launch_bounds__(256)
k_edge_agg(const int* __restrict__ src, const int* __restrict__ dst,
           const float* __restrict__ dinv, const float* __restrict__ h,
           float* __restrict__ agg)
{
    long gid  = (long)blockIdx.x * 256 + threadIdx.x;
    int  e    = (int)(gid >> 5);
    int  lane = (int)(gid & 31);
    if (e >= EE) return;
    int s = src[e], d = dst[e];
    float w = dinv[s] * dinv[d];
    const float4 v = *(const float4*)(h + (size_t)s * FH + lane * 4);
    float* out = agg + (size_t)d * FH + lane * 4;
    atomicAdd(out + 0, v.x * w);
    atomicAdd(out + 1, v.y * w);
    atomicAdd(out + 2, v.z * w);
    atomicAdd(out + 3, v.w * w);
}

// --------------------------------------- self-loop + bias + ReLU + BN stats
__global__ void __launch_bounds__(128)
k_self_bias_relu(const float* __restrict__ h, float* __restrict__ agg,
                 const float* __restrict__ dinv, const float* __restrict__ bias,
                 float* __restrict__ sum, float* __restrict__ sumsq)
{
    int f  = threadIdx.x;
    int i0 = blockIdx.x * 256;
    float bf = bias[f];
    float s = 0.f, ss = 0.f;
    for (int r = 0; r < 256; ++r) {
        int i = i0 + r;
        if (i >= NN) break;
        float di = dinv[i];
        size_t idx = (size_t)i * FH + f;
        float v = agg[idx] + h[idx] * di * di + bf;   // self term = h * 1/deg
        v = fmaxf(v, 0.f);
        agg[idx] = v;
        s += v; ss += v * v;
    }
    atomicAdd(&sum[f], s);
    atomicAdd(&sumsq[f], ss);
}

__global__ void k_bn_finalize(const float* __restrict__ sum, const float* __restrict__ sumsq,
                              const float* __restrict__ gamma, const float* __restrict__ beta,
                              float* __restrict__ scale, float* __restrict__ shift)
{
    int f = threadIdx.x;   // 128 threads, 1 block
    float mu  = sum[f]  * (1.0f / NN);
    float var = sumsq[f] * (1.0f / NN) - mu * mu;
    float sc  = gamma[f] * rsqrtf(var + 1e-5f);
    scale[f] = sc;
    shift[f] = beta[f] - mu * sc;
}

// ------------------------------------------------ pooling (batch is sorted)
__global__ void __launch_bounds__(128)
k_pool(const float* __restrict__ h, const float* __restrict__ scale,
       const float* __restrict__ shift, const int* __restrict__ batch,
       float* __restrict__ pool, float* __restrict__ cnt)
{
    int f  = threadIdx.x;
    int i0 = blockIdx.x * 256;
    float sc = scale[f], sh = shift[f];
    float acc = 0.f, c = 0.f;
    int cur = -1;
    for (int r = 0; r < 256; ++r) {
        int i = i0 + r;
        if (i >= NN) break;
        int g = batch[i];
        if (g != cur) {
            if (cur >= 0) {
                atomicAdd(&pool[(size_t)cur * FH + f], acc);
                if (f == 0) atomicAdd(&cnt[cur], c);
            }
            cur = g; acc = 0.f; c = 0.f;
        }
        acc += h[(size_t)i * FH + f] * sc + sh;
        c   += 1.f;
    }
    if (cur >= 0) {
        atomicAdd(&pool[(size_t)cur * FH + f], acc);
        if (f == 0) atomicAdd(&cnt[cur], c);
    }
}

__global__ void k_fc(const float* __restrict__ pool, const float* __restrict__ cnt,
                     const float* __restrict__ Wfc, const float* __restrict__ bfc,
                     float* __restrict__ out)
{
    int g = blockIdx.x * blockDim.x + threadIdx.x;
    if (g >= GG) return;
    float inv = 1.f / fmaxf(cnt[g], 1.f);
    float a0 = bfc[0], a1 = bfc[1];
    for (int f = 0; f < FH; ++f) {
        float m = pool[(size_t)g * FH + f] * inv;
        a0 += m * Wfc[f * 2 + 0];
        a1 += m * Wfc[f * 2 + 1];
    }
    out[g * 2 + 0] = 1.f / (1.f + expf(-a0));
    out[g * 2 + 1] = 1.f / (1.f + expf(-a1));
}

// ---------------------------------------------------------------- launcher
static inline char* bump(char*& p, size_t bytes) {
    char* r = p;
    p += (bytes + 511) & ~(size_t)511;
    return r;
}

extern "C" void kernel_launch(void* const* d_in, const int* in_sizes, int n_in,
                              void* d_out, int out_size, void* d_ws, size_t ws_size,
                              hipStream_t stream) {
    const float* x      = (const float*)d_in[0];
    const int*   eidx   = (const int*)d_in[1];
    const int*   src    = eidx;
    const int*   dst    = eidx + EE;
    const int*   batch  = (const int*)d_in[2];
    const float* W1     = (const float*)d_in[3];
    const float* b1     = (const float*)d_in[4];
    const float* gamma1 = (const float*)d_in[5];
    const float* beta1  = (const float*)d_in[6];
    const float* W2     = (const float*)d_in[7];
    const float* b2     = (const float*)d_in[8];
    const float* gamma2 = (const float*)d_in[9];
    const float* beta2  = (const float*)d_in[10];
    const float* Wfc    = (const float*)d_in[11];
    const float* bfc    = (const float*)d_in[12];
    float*       out    = (float*)d_out;

    char* p = (char*)d_ws;
    float*  deg   = (float*)bump(p, NN * 4);
    float*  dinv  = (float*)bump(p, NN * 4);
    float*  h     = (float*)bump(p, (size_t)NN * FH * 4);
    float*  agg   = (float*)bump(p, (size_t)NN * FH * 4);
    bf16_t* Wb1   = (bf16_t*)bump(p, 128 * 128 * 2);
    bf16_t* Wb2   = (bf16_t*)bump(p, 128 * 128 * 2);
    float*  stats = (float*)bump(p, 8 * 128 * 4);   // zeroed each call
    float*  sum1 = stats, *sq1 = stats + 128, *sc1 = stats + 256, *sh1 = stats + 384;
    float*  sum2 = stats + 512, *sq2 = stats + 640, *sc2 = stats + 768, *sh2 = stats + 896;
    float*  pool  = (float*)bump(p, (size_t)GG * FH * 4);
    float*  cnt   = (float*)bump(p, GG * 4);

    const int rowBlocks  = NN / ROWS;          // 1250 (exact)
    const int nodeBlocks = (NN + 255) / 256;   // 391
    const long NF = (long)NN * FH;

    // weights -> bf16, fragment-major (deterministic, every call)
    k_convert_w<<<64, 256, 0, stream>>>(W1, Wb1);
    k_convert_w<<<64, 256, 0, stream>>>(W2, Wb2);

    // init workspace state
    k_fill<<<512, 256, 0, stream>>>(deg, NN, 1.0f);          // self-loop
    k_fill<<<2048, 256, 0, stream>>>(agg, NF, 0.0f);
    k_fill<<<4, 256, 0, stream>>>(stats, 8 * 128, 0.0f);
    k_fill<<<256, 256, 0, stream>>>(pool, (long)GG * FH, 0.0f);
    k_fill<<<2, 256, 0, stream>>>(cnt, GG, 0.0f);

    // degrees + D^-1/2
    k_deg<<<(EE + 255) / 256, 256, 0, stream>>>(dst, deg);
    k_dinv<<<nodeBlocks, 256, 0, stream>>>(deg, dinv);

    // ---- layer 1 ----
    k_gemm_bf16<<<rowBlocks, 256, 0, stream>>>(x, Wb1, h, nullptr, nullptr);
    k_edge_agg<<<EE / 8, 256, 0, stream>>>(src, dst, dinv, h, agg);
    k_self_bias_relu<<<nodeBlocks, 128, 0, stream>>>(h, agg, dinv, b1, sum1, sq1);
    k_bn_finalize<<<1, 128, 0, stream>>>(sum1, sq1, gamma1, beta1, sc1, sh1);

    // ---- layer 2 (BN1 fused into GEMM A-load) ----
    k_gemm_bf16<<<rowBlocks, 256, 0, stream>>>(agg, Wb2, h, sc1, sh1);
    k_fill<<<2048, 256, 0, stream>>>(agg, NF, 0.0f);
    k_edge_agg<<<EE / 8, 256, 0, stream>>>(src, dst, dinv, h, agg);
    k_self_bias_relu<<<nodeBlocks, 128, 0, stream>>>(h, agg, dinv, b2, sum2, sq2);
    k_bn_finalize<<<1, 128, 0, stream>>>(sum2, sq2, gamma2, beta2, sc2, sh2);

    // ---- pool (BN2 fused) + FC + sigmoid ----
    k_pool<<<nodeBlocks, 128, 0, stream>>>(agg, sc2, sh2, batch, pool, cnt);
    k_fc<<<(GG + 255) / 256, 256, 0, stream>>>(pool, cnt, Wfc, bfc, out);
}
